// NESDecoder_16569983828588
// MI455X (gfx1250) — matrix operand order repaired and checked by reading, not compiled, not run on hardware
//
#include <hip/hip_runtime.h>
#include <hip/hip_bf16.h>

typedef __attribute__((ext_vector_type(16))) _Float16 v16h;
typedef __attribute__((ext_vector_type(8)))  float    v8f;
typedef __attribute__((ext_vector_type(4)))  unsigned v4u;
typedef __attribute__((ext_vector_type(8)))  int      v8i_;
typedef __attribute__((ext_vector_type(4)))  int      v4i_;

// ---- problem geometry (b = 128) ----
#define NB 128
#define NT 960          // tiles per image (30*32)
#define NN 256          // name count / K dim
#define NP 256          // per-tile pattern vector (8*8*4)
#define NBLK 240        // attribute blocks (15*16)

// ---- d_out float offsets (outputs concatenated in return order) ----
#define IMG_OFF   0                                  // (128,3,240,256)
#define NAMES_OFF (NB*3*240*256)                     // 23,592,960 : (128,960,256)
#define P_OFF     (NAMES_OFF + NB*NT*NN)             // 55,050,240 : (128,256,256)
#define ATTR_OFF  (P_OFF + NB*256*256)               // 63,438,848 : (128,240,8)
#define PAL_OFF   (ATTR_OFF + NB*NBLK*8)             // 63,684,608 : (128,8,12)

// ---- workspace layout ----
#define COLOR_WS_FLOATS ((size_t)NB*NBLK*12)                  // 368,640 f32
#define NAMES_H_BYTES   ((size_t)NB*NT*NN*sizeof(_Float16))   // 62.9 MB

// =====================================================================
// Kernel 1: names softmax.  logits = (names[b,n,t] + gum[b,t,n]) / 2
// one 256-thread WG per (b,t); optional f16 copy to ws for the TDM path.
// =====================================================================
__global__ __launch_bounds__(256)
void names_softmax(const float* __restrict__ names,
                   const float* __restrict__ gum,
                   float* __restrict__ out,
                   _Float16* __restrict__ names_h) {
  int bt = blockIdx.x;
  int b = bt / NT, t = bt % NT;
  int i = threadIdx.x;
  float x = (names[(size_t)b*NN*NT + (size_t)i*NT + t] +
             gum  [(size_t)b*NT*NN + (size_t)t*NN + i]) * 0.5f;
  __shared__ float red[256];
  red[i] = x; __syncthreads();
  for (int s = 128; s > 0; s >>= 1) {
    if (i < s) red[i] = fmaxf(red[i], red[i + s]);
    __syncthreads();
  }
  float m = red[0]; __syncthreads();
  float e = __expf(x - m);
  red[i] = e; __syncthreads();
  for (int s = 128; s > 0; s >>= 1) {
    if (i < s) red[i] += red[i + s];
    __syncthreads();
  }
  float r = e / red[0];
  size_t o = (size_t)b*NT*NN + (size_t)t*NN + i;
  out[NAMES_OFF + o] = r;
  if (names_h) names_h[o] = (_Float16)r;
}

// =====================================================================
// Kernel 2: pattern softmax over 4 gray levels.
// =====================================================================
__global__ __launch_bounds__(256)
void pattern_softmax(const float* __restrict__ pat,
                     float* __restrict__ out) {
  int id = blockIdx.x * 256 + threadIdx.x;        // < 128*256*64 = 2^21
  int b = id >> 14;
  int r = id & 16383;
  int tile = r >> 6, pix = r & 63;
  int d2 = tile >> 4, d4 = tile & 15, d3 = pix >> 3, d5 = pix & 7;
  size_t base = (size_t)b * 65536 + (size_t)(d2*8 + d3) * 128 + (d4*8 + d5);
  float v0 = pat[base];
  float v1 = pat[base + 16384];
  float v2 = pat[base + 2*16384];
  float v3 = pat[base + 3*16384];
  float mx = fmaxf(fmaxf(v0, v1), fmaxf(v2, v3));
  float e0 = __expf(v0 - mx), e1 = __expf(v1 - mx);
  float e2 = __expf(v2 - mx), e3 = __expf(v3 - mx);
  float inv = 1.0f / (e0 + e1 + e2 + e3);
  size_t o = P_OFF + (size_t)b*65536 + (size_t)tile*256 + pix*4;
  out[o+0] = e0*inv; out[o+1] = e1*inv; out[o+2] = e2*inv; out[o+3] = e3*inv;
}

// =====================================================================
// Kernel 3: attribute softmax (over 8 palettes) + (b,240,8)@(b,8,12)
// =====================================================================
__global__ __launch_bounds__(256)
void attr_softmax_color(const float* __restrict__ attr,
                        const float* __restrict__ gumA,
                        const float* __restrict__ pal,
                        float* __restrict__ out,
                        float* __restrict__ color_ws) {
  int id = blockIdx.x * 256 + threadIdx.x;        // < 128*240 = 30720 (exact grid)
  int b = id / NBLK, blk = id % NBLK;
  float v[8], mx = -1e30f;
  for (int e = 0; e < 8; ++e) {
    v[e] = (attr[(size_t)b*8*NBLK + (size_t)e*NBLK + blk] +
            gumA[(size_t)b*NBLK*8 + (size_t)blk*8 + e]) * 0.5f;
    mx = fmaxf(mx, v[e]);
  }
  float s = 0.f;
  for (int e = 0; e < 8; ++e) { v[e] = __expf(v[e] - mx); s += v[e]; }
  float inv = 1.0f / s;
  for (int e = 0; e < 8; ++e) {
    v[e] *= inv;
    out[ATTR_OFF + (size_t)b*NBLK*8 + (size_t)blk*8 + e] = v[e];
  }
  for (int gc = 0; gc < 12; ++gc) {
    float acc = 0.f;
    for (int e = 0; e < 8; ++e) acc += v[e] * pal[(size_t)b*96 + e*12 + gc];
    color_ws[(size_t)b*NBLK*12 + (size_t)blk*12 + gc] = acc;
  }
}

// =====================================================================
// Kernel 4: copy palettes (b,8,12) to output tuple slot 5.
// =====================================================================
__global__ __launch_bounds__(256)
void pal_copy(const float* __restrict__ pal, float* __restrict__ out) {
  int id = blockIdx.x * 256 + threadIdx.x;        // < 128*96 = 12288
  out[PAL_OFF + id] = pal[id];
}

// =====================================================================
// Kernel 5: fused WMMA GEMM + colorize + sigmoid + layout scatter.
//   monochrome(32t x 64pp) = names_idxs(32x256) @ p(256x64)   [f16 WMMA]
// 4 wave32 / WG; each wave owns a 16-col panel and two 16-row halves
// (two interleaved WMMA accumulator chains, 16 wmma per wave).
// USE_TDM: A stripe (f16, from ws) is DMA'd into LDS by the Tensor Data
// Mover, one tensor_load_to_lds per wave, with hardware LDS padding
// reproducing the [row][264] bank-conflict-free stride.
// =====================================================================
template <bool USE_TDM>
__global__ __launch_bounds__(128)
void gemm_fused(const float* __restrict__ dout_ro,   // names_idxs / p regions
                const _Float16* __restrict__ names_h,
                const float* __restrict__ color_ws,
                float* __restrict__ out) {
  const int panel = blockIdx.x;        // 0..3  -> pp0 = panel*64
  const int tileY = blockIdx.y;        // 0..29 -> t0 = tileY*32
  const int b     = blockIdx.z;        // 0..127
  const int t0  = tileY * 32;
  const int pp0 = panel * 64;
  const int tid  = threadIdx.x;
  const int lane = tid & 31;
  const int wave = tid >> 5;

  __shared__ _Float16 lA[32][264];       // A stripe, [m][k], k-contiguous
  __shared__ _Float16 lB[64][264];       // B panel, [n_local][k], k-contiguous
  __shared__ float    lC[4][2][16][16];  // per-wave accum tiles (2 row-halves)

  const float* Bg = dout_ro + P_OFF + (size_t)b*NN*NP + pp0;

  if (USE_TDM) {
    // --- A via Tensor Data Mover: each wave DMAs its 8-row f16 slice ---
    int w = __builtin_amdgcn_readfirstlane(wave);
    unsigned lds_base = (unsigned)(uintptr_t)(&lA[0][0]) + (unsigned)(w * 8 * 264 * 2);
    unsigned long long ga =
        (unsigned long long)(uintptr_t)(names_h + (size_t)b*NT*NN + (size_t)t0*NN + w*2048);
    v4u g0;
    g0[0] = 1u;                                   // count=1 (valid descriptor)
    g0[1] = lds_base;                             // lds_addr
    g0[2] = (unsigned)(ga & 0xFFFFFFFFu);         // global_addr[31:0]
    g0[3] = (unsigned)((ga >> 32) & 0x1FFFFFFu)   // global_addr[56:32]
          | (2u << 30);                           // type=2 ("image")
    v8i_ g1;
    g1[0] = (int)((1u << 16)                      // data_size = 2 bytes
                | (1u << 20)                      // pad_enable
                | (6u << 22)                      // pad_interval: every 128 DWORDs
                | (3u << 25));                    // pad_amount: 4 DWORDs (stride 264 h)
    g1[1] = (int)(2048u << 16);                   // tensor_dim0 = 2048 (low16 @ [31:16])
    g1[2] = (int)(1u << 16);                      // tensor_dim1 = 1
    g1[3] = (int)(2048u << 16);                   // tile_dim0 = 2048
    g1[4] = 0;                                    // tile_dim1/2 unused
    g1[5] = 2048;                                 // tensor_dim0_stride = 2048
    g1[6] = 0; g1[7] = 0;
    v4i_ gz4 = {0, 0, 0, 0};
    v8i_ gz8 = {0, 0, 0, 0, 0, 0, 0, 0};
    // amdgpu-toolchain (clang-23 / therock-10.0) 6-arg form
    __builtin_amdgcn_tensor_load_to_lds(g0, g1, gz4, gz4, gz8, 0);
  } else {
    // --- A staged from fp32 names_idxs with in-flight f32->f16 convert ---
    const float* Ag = dout_ro + NAMES_OFF + (size_t)b*NT*NN + (size_t)t0*NN;
    for (int idx4 = tid; idx4 < 2048; idx4 += 128) {   // 32x256 / 4
      int m = idx4 >> 6, k4 = (idx4 & 63) * 4;
      float4 v = *(const float4*)(Ag + (size_t)m*NN + k4);
      lA[m][k4+0] = (_Float16)v.x; lA[m][k4+1] = (_Float16)v.y;
      lA[m][k4+2] = (_Float16)v.z; lA[m][k4+3] = (_Float16)v.w;
    }
  }

  // --- B staged transposed: p[n][pp0+j] -> lB[j][n], f32 -> f16 ---
  for (int idx4 = tid; idx4 < 4096; idx4 += 128) {
    int n = idx4 >> 4, j4 = (idx4 & 15) * 4;
    float4 v = *(const float4*)(Bg + (size_t)n*NP + j4);
    lB[j4+0][n] = (_Float16)v.x; lB[j4+1][n] = (_Float16)v.y;
    lB[j4+2][n] = (_Float16)v.z; lB[j4+3][n] = (_Float16)v.w;
  }
  if (USE_TDM) __builtin_amdgcn_s_wait_tensorcnt(0);
  __syncthreads();

  // --- WMMA: 8 K-steps, two row-halves per wave (interleaved chains) ---
  const int m_a   = lane & 15;
  const int abase = (lane >= 16) ? 8 : 0;   // A lane K-offset (ISA 16-bit A layout)
  const int nloc  = wave * 16 + (lane & 15);
  const int bkoff = (lane >= 16) ? 16 : 0;  // unused; B is read with abase, same k order as A
  v8f c0 = {}, c1 = {};
#pragma unroll
  for (int kk = 0; kk < 8; ++kk) {
    v16h a0, a1, bb;
    const int kbase = kk * 32;
#pragma unroll
    for (int v = 0; v < 8; ++v) {
      int ka = kbase + ((v < 4) ? 0 : 16) + abase + (v & 3) * 2;
      a0[2*v+0] = lA[m_a][ka+0];      a0[2*v+1] = lA[m_a][ka+1];
      a1[2*v+0] = lA[m_a + 16][ka+0]; a1[2*v+1] = lA[m_a + 16][ka+1];
      int kb = kbase + ((v < 4) ? 0 : 16) + abase + (v & 3) * 2;
      bb[2*v+0] = lB[nloc][kb+0];     bb[2*v+1] = lB[nloc][kb+1];
    }
    c0 = __builtin_amdgcn_wmma_f32_16x16x32_f16(false, a0, false, bb,
                                                (short)0, c0, false, false);
    c1 = __builtin_amdgcn_wmma_f32_16x16x32_f16(false, a1, false, bb,
                                                (short)0, c1, false, false);
  }

  // spill per ISA C/D layout: lane col = lane&15, row = v + 8*(lane>=16)
#pragma unroll
  for (int v = 0; v < 8; ++v) {
    int rr = v + ((lane >= 16) ? 8 : 0);
    lC[wave][0][rr][lane & 15] = c0[v];
    lC[wave][1][rr][lane & 15] = c1[v];
  }
  __syncthreads();

  // --- epilogue: 512 (t,pix) pairs, 4 per thread; 3 channels each ---
  // q decode is pure shift/mask: t_local = q>>4, pix_local = q&15.
  const int pixBase = pp0 >> 2;
  const size_t imgB = (size_t)b * (3*240*256);
#pragma unroll
  for (int rep = 0; rep < 4; ++rep) {
    int q = tid + rep * 128;
    int t_local = q >> 4;
    int pix_local = q & 15;
    int wv = pix_local >> 2;                 // wave tile holding these 4 grays
    int cbase = (pix_local * 4) & 15;
    int th = t_local >> 4, tr = t_local & 15;
    // 4 gray values loaded once, reused by all 3 channels
    float m0 = lC[wv][th][tr][cbase + 0];
    float m1 = lC[wv][th][tr][cbase + 1];
    float m2 = lC[wv][th][tr][cbase + 2];
    float m3 = lC[wv][th][tr][cbase + 3];
    int t  = t0 + t_local;
    int ty = t >> 5, tx = t & 31;
    int blk = (ty >> 1) * 16 + (tx >> 1);
    const float* col = color_ws + (size_t)b*NBLK*12 + (size_t)blk*12;
    int pix = pixBase + pix_local;
    int py = pix >> 3, px = pix & 7;
    size_t obase = imgB + (size_t)(ty*8 + py)*256 + (tx*8 + px);
#pragma unroll
    for (int cch = 0; cch < 3; ++cch) {
      float acc = m0 * col[0*3 + cch] + m1 * col[1*3 + cch]
                + m2 * col[2*3 + cch] + m3 * col[3*3 + cch];
      out[IMG_OFF + obase + (size_t)cch*(240*256)] =
          1.0f / (1.0f + __expf(-acc));
    }
  }
}

// =====================================================================
extern "C" void kernel_launch(void* const* d_in, const int* in_sizes, int n_in,
                              void* d_out, int out_size, void* d_ws, size_t ws_size,
                              hipStream_t stream) {
  const float* names    = (const float*)d_in[0];
  const float* patterns = (const float*)d_in[1];
  const float* attrs    = (const float*)d_in[2];
  const float* palettes = (const float*)d_in[3];
  const float* gum_n    = (const float*)d_in[4];
  const float* gum_a    = (const float*)d_in[5];
  float* out = (float*)d_out;

  float* color_ws = (float*)d_ws;                       // 1.47 MB
  size_t color_bytes = COLOR_WS_FLOATS * sizeof(float);
  bool use_tdm = ws_size >= color_bytes + NAMES_H_BYTES;
  _Float16* names_h = use_tdm ? (_Float16*)((char*)d_ws + color_bytes) : nullptr;

  names_softmax<<<dim3(NB * NT), dim3(256), 0, stream>>>(names, gum_n, out, names_h);
  pattern_softmax<<<dim3((NB * 256 * 64) / 256), dim3(256), 0, stream>>>(patterns, out);
  attr_softmax_color<<<dim3((NB * NBLK) / 256), dim3(256), 0, stream>>>(
      attrs, gum_a, palettes, out, color_ws);
  pal_copy<<<dim3((NB * 96) / 256), dim3(256), 0, stream>>>(palettes, out);
  if (use_tdm)
    gemm_fused<true><<<dim3(4, 30, NB), dim3(128), 0, stream>>>(out, names_h, color_ws, out);
  else
    gemm_fused<false><<<dim3(4, 30, NB), dim3(128), 0, stream>>>(out, names_h, color_ws, out);
}
